// gcn_layer_v2_7404523619090
// MI455X (gfx1250) — matrix operand order repaired and checked by reading, not compiled, not run on hardware
//
#include <hip/hip_runtime.h>

typedef __attribute__((ext_vector_type(16))) __bf16 v16bf;
typedef __attribute__((ext_vector_type(8)))  float  v8f;

#define B_ 16
#define L_ 4096
#define D_ 256
#define ROWS_PER_WG 128
#define LDSTRIDE 264   // padded row stride (ushorts) to spread LDS banks

__device__ __forceinline__ unsigned short f32_to_bf16_rne(float f) {
  unsigned u = __float_as_uint(f);
  unsigned r = u + 0x7FFFu + ((u >> 16) & 1u);
  return (unsigned short)(r >> 16);
}
__device__ __forceinline__ float bf16_to_f32(unsigned short h) {
  return __uint_as_float(((unsigned)h) << 16);
}
__device__ __forceinline__ float sigmoidf_(float x) {
  return 1.0f / (1.0f + __expf(-x));
}

// ---------------- zero child-scatter accumulator ----------------
__global__ void k_zero(float4* __restrict__ p) {
  p[(size_t)blockIdx.x * 256 + threadIdx.x] = make_float4(0.f, 0.f, 0.f, 0.f);
}

// ---------------- convert W_loop|W_child|W_parent to bf16, transposed [m][n][k] ----------------
__global__ void k_cvt(const float* __restrict__ Wl, const float* __restrict__ Wc,
                      const float* __restrict__ Wp, unsigned short* __restrict__ WT) {
  int idx = blockIdx.x * 256 + threadIdx.x;     // 0 .. 3*256*256-1
  int m   = idx >> 16;
  int rem = idx & 65535;
  int n   = rem >> 8;
  int k   = rem & 255;
  const float* W = (m == 0) ? Wl : ((m == 1) ? Wc : Wp);
  WT[idx] = f32_to_bf16_rne(W[k * D_ + n]);     // WT[(m*256+n)*256+k] == idx
}

// B tile pointer for N-tile `nt` (column per lane, K = k0+8h..+7 and k0+16+8h..+7 per half-wave)
__device__ __forceinline__ const unsigned short* btile_ptr(
    const unsigned short* __restrict__ WT, int nt, int r16, int half) {
  const int m = nt >> 4, nb = nt & 15;
  const int c = nb * 16 + r16;
  return WT + ((size_t)(m * D_ + c) * D_) + 8 * half;
}

__device__ __forceinline__ void load_btile(const unsigned short* __restrict__ p,
                                           v16bf* __restrict__ dst) {
  #pragma unroll
  for (int ks = 0; ks < 8; ++ks) {
    union { uint4 q[2]; v16bf v; } u;
    u.q[0] = *(const uint4*)(p + 32 * ks);
    u.q[1] = *(const uint4*)(p + 32 * ks + 16);
    dst[ks] = u.v;
  }
}

// fused epilogue; C layout: VGPR r -> M = r + 8*(lane/16), N = lane%16
__device__ __forceinline__ void do_epilogue(
    int nt, v8f acc, int wave, int half, int r16,
    size_t row0, size_t bbase,
    const float* __restrict__ b_loop, const float* __restrict__ b_child,
    const float* __restrict__ b_parent, const float* __restrict__ E_cw,
    const float* gl_lds, const float* gc_lds,
    const int* dep_lds, const int* mpp_lds,
    float* __restrict__ outLoop, float* __restrict__ ACCc, float* __restrict__ Z)
{
  const int m  = nt >> 4;
  const int nb = nt & 15;
  const int c  = nb * 16 + r16;
  #pragma unroll
  for (int r = 0; r < 8; ++r) {
    const int riw = 16 * wave + 8 * half + r;
    const size_t row = row0 + riw;
    const float v = acc[r];
    if (m == 0) {
      outLoop[row * D_ + c] = gl_lds[riw] * (v + b_loop[c]);
    } else if (m == 1) {
      float cv = gc_lds[riw] * (v + b_child[c] + E_cw[(size_t)dep_lds[riw] * D_ + c]);
      unsafeAtomicAdd(&ACCc[(bbase + (size_t)mpp_lds[riw]) * D_ + c], cv);
    } else {
      Z[row * D_ + c] = v + b_parent[c];
    }
  }
}

// ---------------- fused triple GEMM + gates + epilogues ----------------
__global__ __launch_bounds__(256) void k_gemm(
    const float* __restrict__ x,
    const int*   __restrict__ ppos,
    const int*   __restrict__ deps,
    const int*   __restrict__ amask,
    const float* __restrict__ b_loop,
    const float* __restrict__ b_child,
    const float* __restrict__ b_parent,
    const float* __restrict__ wlg, const float* __restrict__ blg,
    const float* __restrict__ wcg, const float* __restrict__ bcg,
    const float* __restrict__ wpg, const float* __restrict__ bpg,
    const float* __restrict__ E_cw,
    const float* __restrict__ E_pg, const float* __restrict__ E_cg,
    const unsigned short* __restrict__ WT,
    float* __restrict__ Z, float* __restrict__ ACCc,
    float* __restrict__ PG, float* __restrict__ outLoop)
{
  __shared__ unsigned short a_lds[ROWS_PER_WG * LDSTRIDE];
  __shared__ float gl_lds[ROWS_PER_WG];
  __shared__ float gc_lds[ROWS_PER_WG];
  __shared__ int   dep_lds[ROWS_PER_WG];
  __shared__ int   mpp_lds[ROWS_PER_WG];

  const int tid = threadIdx.x;
  const size_t row0 = (size_t)blockIdx.x * ROWS_PER_WG;   // flat row over B*L

  // ---- stage x rows into LDS as bf16 ----
  const float4* xsrc = (const float4*)(x + row0 * D_);
  #pragma unroll
  for (int j = 0; j < 32; ++j) {
    int i = tid + 256 * j;                 // float4 index, 0..8191
    float4 v = xsrc[i];
    ushort4 h;
    h.x = f32_to_bf16_rne(v.x);
    h.y = f32_to_bf16_rne(v.y);
    h.z = f32_to_bf16_rne(v.z);
    h.w = f32_to_bf16_rne(v.w);
    int row = i >> 6;
    int kq  = (i & 63) << 2;
    *(ushort4*)&a_lds[row * LDSTRIDE + kq] = h;
  }
  __syncthreads();

  // ---- per-row gates (GEMV from LDS bf16, fp32 accumulate) ----
  if (tid < ROWS_PER_WG) {
    const int row = tid;
    const size_t flat = row0 + row;
    const int dep = deps[flat];
    float dl = 0.f, dc = 0.f, dp = 0.f;
    const unsigned short* ar = &a_lds[row * LDSTRIDE];
    for (int k = 0; k < D_; ++k) {
      float xv = bf16_to_f32(ar[k]);
      dl = fmaf(xv, wlg[k], dl);
      dc = fmaf(xv, wcg[k], dc);
      dp = fmaf(xv, wpg[k], dp);
    }
    gl_lds[row] = sigmoidf_(dl + blg[0]);
    gc_lds[row] = sigmoidf_(dc + bcg[0] + E_cg[dep]);
    PG[flat]    = sigmoidf_(dp + bpg[0] + E_pg[dep]);
    dep_lds[row] = dep;
    mpp_lds[row] = (amask[flat] != 0) ? ppos[flat] : 0;
  }
  __syncthreads();

  const int wave = tid >> 5;
  const int lane = tid & 31;
  const int half = lane >> 4;
  const int r16  = lane & 15;

  // ---- hoist all 8 A fragments (K = 0..255) for this wave's 16-row tile ----
  // ISA layout: lanes 0-15 row M=lane, elems 0..7 -> K = k0+8h..+7, elems 8..15 -> K = k0+16+8h..+7
  v16bf af[8];
  const unsigned short* arow = &a_lds[(16 * wave + r16) * LDSTRIDE];
  #pragma unroll
  for (int ks = 0; ks < 8; ++ks) {
    const unsigned short* p = arow + 32 * ks + 8 * half;
    union { uint4 q[2]; v16bf v; } u;
    u.q[0] = *(const uint4*)(p);        // K = k0+8h .. +7
    u.q[1] = *(const uint4*)(p + 16);   // K = k0+16+8h .. +7
    af[ks] = u.v;
  }

  const size_t bbase = row0 & ~(size_t)(L_ - 1);   // bIdx * L

  // ---- 48 N-tiles: [0..15]=loop, [16..31]=child, [32..47]=parent ----
  // Ping-pong B buffers: tile nt+1 is loaded before tile nt's WMMA chain runs,
  // so every B load has a full 8-WMMA chain (+ epilogue) of latency cover.
  v16bf b0[8], b1[8];
  load_btile(btile_ptr(WT, 0, r16, half), b0);
  #pragma unroll 1
  for (int nt = 0; nt < 48; nt += 2) {
    load_btile(btile_ptr(WT, nt + 1, r16, half), b1);
    {
      v8f acc = {};
      #pragma unroll
      for (int ks = 0; ks < 8; ++ks)
        acc = __builtin_amdgcn_wmma_f32_16x16x32_bf16(
            false, af[ks], false, b0[ks], (short)0, acc, false, false);
      do_epilogue(nt, acc, wave, half, r16, row0, bbase,
                  b_loop, b_child, b_parent, E_cw,
                  gl_lds, gc_lds, dep_lds, mpp_lds, outLoop, ACCc, Z);
    }
    if (nt + 2 < 48)
      load_btile(btile_ptr(WT, nt + 2, r16, half), b0);
    {
      v8f acc = {};
      #pragma unroll
      for (int ks = 0; ks < 8; ++ks)
        acc = __builtin_amdgcn_wmma_f32_16x16x32_bf16(
            false, af[ks], false, b1[ks], (short)0, acc, false, false);
      do_epilogue(nt + 1, acc, wave, half, r16, row0, bbase,
                  b_loop, b_child, b_parent, E_cw,
                  gl_lds, gc_lds, dep_lds, mpp_lds, outLoop, ACCc, Z);
    }
  }
}

// ---------------- final fuse: loop + gated parent gather + child scatter result, mask ----------------
__global__ void k_final(const float* __restrict__ Z, const float* __restrict__ ACCc,
                        const float* __restrict__ PG,
                        const int* __restrict__ ppos, const int* __restrict__ deps,
                        const int* __restrict__ amask, const float* __restrict__ E_pw,
                        float* out)
{
  const int t   = blockIdx.x * 256 + threadIdx.x;  // over B*L*64 float4s
  const int row = t >> 6;
  const int c   = (t & 63) << 2;
  float4 res = make_float4(0.f, 0.f, 0.f, 0.f);
  if (amask[row] != 0) {
    const int bb  = row >> 12;
    const int pp  = ppos[row];
    const int dep = deps[row];
    const float pg = PG[row];
    const float4 zv = *(const float4*)(Z + ((size_t)((bb << 12) + pp)) * D_ + c);
    const float4 ew = *(const float4*)(E_pw + (size_t)dep * D_ + c);
    const float4 lp = *(const float4*)(out + (size_t)row * D_ + c);
    const float4 ch = *(const float4*)(ACCc + (size_t)row * D_ + c);
    res.x = lp.x + pg * (zv.x + ew.x) + ch.x;
    res.y = lp.y + pg * (zv.y + ew.y) + ch.y;
    res.z = lp.z + pg * (zv.z + ew.z) + ch.z;
    res.w = lp.w + pg * (zv.w + ew.w) + ch.w;
  }
  *(float4*)(out + (size_t)row * D_ + c) = res;
}

extern "C" void kernel_launch(void* const* d_in, const int* in_sizes, int n_in,
                              void* d_out, int out_size, void* d_ws, size_t ws_size,
                              hipStream_t stream) {
  (void)in_sizes; (void)n_in; (void)out_size; (void)ws_size;
  const float* x       = (const float*)d_in[0];
  const int*   ppos    = (const int*)d_in[1];
  const int*   deps    = (const int*)d_in[2];
  const int*   amask   = (const int*)d_in[3];
  const float* W_loop  = (const float*)d_in[4];
  const float* b_loop  = (const float*)d_in[5];
  const float* W_child = (const float*)d_in[6];
  const float* b_child = (const float*)d_in[7];
  const float* W_par   = (const float*)d_in[8];
  const float* b_par   = (const float*)d_in[9];
  const float* wlg = (const float*)d_in[10];
  const float* blg = (const float*)d_in[11];
  const float* wcg = (const float*)d_in[12];
  const float* bcg = (const float*)d_in[13];
  const float* wpg = (const float*)d_in[14];
  const float* bpg = (const float*)d_in[15];
  const float* E_pw = (const float*)d_in[16];
  const float* E_cw = (const float*)d_in[17];
  const float* E_pg = (const float*)d_in[18];
  const float* E_cg = (const float*)d_in[19];

  char* ws = (char*)d_ws;
  const size_t SZ = (size_t)B_ * L_ * D_ * sizeof(float);    // 64 MB
  float* Z    = (float*)(ws);
  float* ACCc = (float*)(ws + SZ);
  float* PG   = (float*)(ws + 2 * SZ);
  unsigned short* WT = (unsigned short*)(ws + 2 * SZ + (size_t)B_ * L_ * sizeof(float));
  float* out = (float*)d_out;

  k_zero<<<(B_ * L_ * D_ / 4) / 256, 256, 0, stream>>>((float4*)ACCc);
  k_cvt<<<(3 * D_ * D_) / 256, 256, 0, stream>>>(W_loop, W_child, W_par, WT);
  k_gemm<<<(B_ * L_) / ROWS_PER_WG, 256, 0, stream>>>(
      x, ppos, deps, amask, b_loop, b_child, b_par,
      wlg, blg, wcg, bcg, wpg, bpg, E_cw, E_pg, E_cg,
      WT, Z, ACCc, PG, out);
  k_final<<<(B_ * L_ * D_ / 4) / 256, 256, 0, stream>>>(
      Z, ACCc, PG, ppos, deps, amask, E_pw, out);
}
